// NNFOwithBayesianJumps_45097156608134
// MI455X (gfx1250) — matrix operand order repaired and checked by reading, not compiled, not run on hardware
//
#include <hip/hip_runtime.h>
#include <hip/hip_bf16.h>

typedef __attribute__((ext_vector_type(16))) _Float16 v16h;
typedef __attribute__((ext_vector_type(8)))  _Float16 v8h;
typedef __attribute__((ext_vector_type(8)))  float    v8f;

#define NBLK   16
#define NTHR   256
#define NWAVES (NBLK * (NTHR / 32))   // 128 waves
#define NTOT   (NBLK * NTHR)          // 4096 threads

#define Hh   256
#define Dd   64
#define Pp   10
#define DPc  640
#define Bb   256
#define TSs  600
#define DTc      0.05f
#define LOG2PI   1.8378770664093453f    /* 2*log(sqrt(2pi)) */
#define LOG_OBS  (-4.605170185988091f)  /* log(0.01) */
#define INV2VAR  5000.0f                /* 1/(2*0.01^2) */

struct KParams {
  // inputs (setup_inputs order)
  const float *X, *Mm, *cov, *cm_w1, *cm_b1, *cm_w2, *cm_b2;
  const float *pm_w1, *pm_b1, *pm_w2, *pm_b2;
  const float *gc_wxz, *gc_bxz, *gc_wxn, *gc_bxn, *gc_whz, *gc_whn;
  const float *wp, *bp;
  const float *gru_wih, *gru_whh, *gru_bih, *gru_bhh;
  const float *cls_w1, *cls_b1, *cls_w2, *cls_b2;
  // workspace-derived
  int*   bar;        // [2] arrive / generation
  float* partial;    // [NBLK] loss partials
  _Float16 *W1, *W2, *WZ, *WN, *WIH, *WHH;       // packed f16 weights
  float*    h;                                    // f32 master state [B][H]
  _Float16 *h16, *t116, *p16, *zh16, *gi16;       // f16 activations
  float    *pbuf, *zg, *GA, *GB, *hid1;           // f32 activations
  float*    out;                                  // d_out
};

__device__ __forceinline__ float sigm(float x) { return 1.0f / (1.0f + expf(-x)); }

// ---------------------------------------------------------------------------
// Grid barrier: cluster-barrier fast path (NOP/hw-barrier) + atomic fallback.
// ---------------------------------------------------------------------------
__device__ void grid_barrier(int* bar) {
  __syncthreads();
  if (threadIdx.x == 0) {
    if (__builtin_amdgcn_cluster_id_x() != 0)
      __builtin_amdgcn_s_cluster_barrier();   // hw cluster sync when cluster-dispatched
    int gen = __hip_atomic_load(bar + 1, __ATOMIC_RELAXED, __HIP_MEMORY_SCOPE_AGENT);
    if (__hip_atomic_fetch_add(bar, 1, __ATOMIC_ACQ_REL, __HIP_MEMORY_SCOPE_AGENT) == NBLK - 1) {
      __hip_atomic_store(bar, 0, __ATOMIC_RELAXED, __HIP_MEMORY_SCOPE_AGENT);
      __hip_atomic_fetch_add(bar + 1, 1, __ATOMIC_RELEASE, __HIP_MEMORY_SCOPE_AGENT);
    } else {
      while (__hip_atomic_load(bar + 1, __ATOMIC_ACQUIRE, __HIP_MEMORY_SCOPE_AGENT) == gen)
        __builtin_amdgcn_s_sleep(2);
    }
  }
  __syncthreads();
}

// ---------------------------------------------------------------------------
// One 16x16 output tile, accumulating ktCnt K-slices of 32 via WMMA f16.
//   A: row-major f16 [*, lda]; Wp: packed B fragments, per (nt,kt) tile:
//   lane-major, 16 contiguous halves per lane (32B/lane coalesced loads).
// ---------------------------------------------------------------------------
__device__ __forceinline__ v8f wmma_part(v8f c, const _Float16* __restrict__ A, int lda,
                                         const _Float16* __restrict__ Wp, int ktTot, int nt,
                                         int ktBeg, int ktCnt, int m0) {
  const int lane = threadIdx.x & 31;
  const _Float16* arow = A + (size_t)(m0 + (lane & 15)) * lda + (lane >> 4) * 8;
  const _Float16* wp   = Wp + ((size_t)(nt * ktTot + ktBeg) * 32 + lane) * 16;
  for (int kt = 0; kt < ktCnt; ++kt) {
    v8h a0 = *(const v8h*)(arow);        // K = base + half*8 + 0..7
    v8h a1 = *(const v8h*)(arow + 16);   // K = base + 16 + half*8 + 0..7
    v16h a = __builtin_shufflevector(a0, a1, 0,1,2,3,4,5,6,7,8,9,10,11,12,13,14,15);
    v16h b = *(const v16h*)(wp);
    c = __builtin_amdgcn_wmma_f32_16x16x32_f16(false, a, false, b, (short)0, c, false, false);
    arow += 32;
    wp   += 512;
  }
  return c;
}

// Pack W[N][K] (f32, torch Linear layout) -> f16 WMMA-B fragment order.
__device__ void pack1(const float* __restrict__ W, _Float16* __restrict__ dst,
                      int K, int N, int gtid) {
  const int Kt = K >> 5;
  const int total = K * N;
  for (int idx = gtid; idx < total; idx += NTOT) {
    int i    = idx & 15;
    int lane = (idx >> 4) & 31;
    int blk  = idx >> 9;            // nt*Kt + kt
    int kt = blk % Kt, nt = blk / Kt;
    int k = kt * 32 + (lane >> 4) * 8 + (i & 7) + (i >> 3) * 16;
    int n = nt * 16 + (lane & 15);
    dst[idx] = (_Float16)W[(size_t)n * K + k];
  }
}

// Pack K-concatenated [Wa | Wb] (both [N][Ka], [N][Kb]) -> f16 fragments.
__device__ void pack2(const float* __restrict__ Wa, int Ka,
                      const float* __restrict__ Wb, int Kb,
                      _Float16* __restrict__ dst, int N, int gtid) {
  const int K = Ka + Kb;
  const int Kt = K >> 5;
  const int total = K * N;
  for (int idx = gtid; idx < total; idx += NTOT) {
    int i    = idx & 15;
    int lane = (idx >> 4) & 31;
    int blk  = idx >> 9;
    int kt = blk % Kt, nt = blk / Kt;
    int k = kt * 32 + (lane >> 4) * 8 + (i & 7) + (i >> 3) * 16;
    int n = nt * 16 + (lane & 15);
    float v = (k < Ka) ? Wa[(size_t)n * Ka + k] : Wb[(size_t)n * Kb + (k - Ka)];
    dst[idx] = (_Float16)v;
  }
}

__global__ void nnfo_init(int* bar) { bar[0] = 0; bar[1] = 0; }

// ---------------------------------------------------------------------------
// Persistent cooperative kernel: whole scan in one launch.
// ---------------------------------------------------------------------------
__global__ __launch_bounds__(NTHR) void nnfo_persistent(KParams kp) {
  const int tid  = threadIdx.x;
  const int gtid = blockIdx.x * NTHR + tid;
  const int wid  = blockIdx.x * (NTHR / 32) + (tid >> 5);
  const int lane = tid & 31;
  const int lh8  = (lane >> 4) * 8;   // C-row offset for this half-wave
  const int l16  = lane & 15;         // C-column within tile
  __shared__ float sred[NTHR];

  float accL1 = 0.0f, accL2 = 0.0f;

  // ---- Prologue A: pack all weights to f16 fragments + covariates layer 1
  pack1(kp.pm_w1, kp.W1, 256, 256, gtid);
  pack1(kp.pm_w2, kp.W2, 256, 128, gtid);
  pack2(kp.gc_wxz, 128, kp.gc_whz, 256, kp.WZ, 256, gtid);
  pack2(kp.gc_wxn, 128, kp.gc_whn, 256, kp.WN, 256, gtid);
  pack1(kp.gru_wih, kp.WIH, 640, 768, gtid);
  pack1(kp.gru_whh, kp.WHH, 256, 768, gtid);
  for (int i = gtid; i < Bb * 64; i += NTOT) {           // hid1 = relu(cov@cm_w1^T+b1)
    int b = i >> 6, c = i & 63;
    float acc = kp.cm_b1[c];
    for (int k = 0; k < 16; ++k) acc += kp.cov[b * 16 + k] * kp.cm_w1[c * 16 + k];
    kp.hid1[i] = acc > 0.0f ? acc : 0.0f;
  }
  grid_barrier(kp.bar);

  // ---- Prologue B: h0 = tanh(hid1@cm_w2^T+b2)
  for (int i = gtid; i < Bb * Hh; i += NTOT) {
    int b = i >> 8, o = i & 255;
    float acc = kp.cm_b2[o];
    for (int j = 0; j < 64; ++j) acc += kp.hid1[b * 64 + j] * kp.cm_w2[o * 64 + j];
    float hv = tanhf(acc);
    kp.h[i] = hv;
    kp.h16[i] = (_Float16)hv;
  }
  grid_barrier(kp.bar);

  // ================= main scan =================
  for (int t = 0; t < TSs; ++t) {
    const float* Xt = kp.X  + (size_t)t * Bb * Dd;
    const float* Mt = kp.Mm + (size_t)t * Bb * Dd;

    // fused: KL (l2) of previous step, reading pbuf from previous P11
    if (t > 0) {
      const float* Xp = Xt - Bb * Dd;
      const float* Mp = Mt - Bb * Dd;
      for (int i = gtid; i < Bb * Dd; i += NTOT) {
        int b = i >> 6, d = i & 63;
        float mean2 = kp.pbuf[b * 128 + d];
        float lv2   = kp.pbuf[b * 128 + 64 + d];
        float dx = mean2 - Xp[i];
        float kl = LOG_OBS - 0.5f * lv2 + (expf(lv2) + dx * dx) * INV2VAR - 0.5f;
        accL2 += kl * Mp[i];
      }
    }

    // speculative prefetch of next step's HBM-resident X/M slices
    // (global_prefetch_b8; a full step of GEMM work hides the fetch latency)
    if (t + 1 < TSs) {
      const float* Xn = kp.X  + (size_t)(t + 1) * Bb * Dd;
      const float* Mn = kp.Mm + (size_t)(t + 1) * Bb * Dd;
      for (int i = gtid; i < Bb * Dd; i += NTOT) {
        __builtin_prefetch(Xn + i, 0, 1);
        __builtin_prefetch(Mn + i, 0, 1);
      }
    }

    // P1: t1 = relu(h @ pm_w1^T + b)    [256x256, K=256]
    for (int tt = wid; tt < 256; tt += NWAVES) {
      int mt = tt >> 4, nt = tt & 15;
      v8f c = {};
      c = wmma_part(c, kp.h16, 256, kp.W1, 8, nt, 0, 8, mt * 16);
      int n = nt * 16 + l16;
      float bv = kp.pm_b1[n];
      _Float16* dst = kp.t116 + (size_t)(mt * 16 + lh8) * 256 + n;
      #pragma unroll
      for (int r = 0; r < 8; ++r) {
        float v = c[r] + bv;
        dst[(size_t)r * 256] = (_Float16)(v > 0.0f ? v : 0.0f);
      }
    }
    grid_barrier(kp.bar);

    // P2: p = t1 @ pm_w2^T + b -> p16   [256x128, K=256]
    for (int tt = wid; tt < 128; tt += NWAVES) {
      int mt = tt >> 3, nt = tt & 7;
      v8f c = {};
      c = wmma_part(c, kp.t116, 256, kp.W2, 8, nt, 0, 8, mt * 16);
      int n = nt * 16 + l16;
      float bv = kp.pm_b2[n];
      _Float16* dh = kp.p16 + (size_t)(mt * 16 + lh8) * 128 + n;
      #pragma unroll
      for (int r = 0; r < 8; ++r) dh[(size_t)r * 128] = (_Float16)(c[r] + bv);
    }
    grid_barrier(kp.bar);

    // P3: zg = sigmoid([p|h] @ WZ + bxz); zh16 = f16(zg*h)   [K=384]
    for (int tt = wid; tt < 256; tt += NWAVES) {
      int mt = tt >> 4, nt = tt & 15, m0 = mt * 16;
      v8f c = {};
      c = wmma_part(c, kp.p16, 128, kp.WZ, 12, nt, 0, 4, m0);
      c = wmma_part(c, kp.h16, 256, kp.WZ, 12, nt, 4, 8, m0);
      int n = nt * 16 + l16;
      float bv = kp.gc_bxz[n];
      #pragma unroll
      for (int r = 0; r < 8; ++r) {
        size_t idx = (size_t)(m0 + lh8 + r) * 256 + n;
        float zv = sigm(c[r] + bv);
        kp.zg[idx] = zv;
        kp.zh16[idx] = (_Float16)(zv * kp.h[idx]);
      }
    }
    grid_barrier(kp.bar);

    // P4: n = tanh([p|zg*h] @ WN + bxn); h += dt*(1-z)*(n-h)   [K=384]
    for (int tt = wid; tt < 256; tt += NWAVES) {
      int mt = tt >> 4, nt = tt & 15, m0 = mt * 16;
      v8f c = {};
      c = wmma_part(c, kp.p16,  128, kp.WN, 12, nt, 0, 4, m0);
      c = wmma_part(c, kp.zh16, 256, kp.WN, 12, nt, 4, 8, m0);
      int n = nt * 16 + l16;
      float bv = kp.gc_bxn[n];
      #pragma unroll
      for (int r = 0; r < 8; ++r) {
        size_t idx = (size_t)(m0 + lh8 + r) * 256 + n;
        float nv = tanhf(c[r] + bv);
        float zv = kp.zg[idx];
        float hv = kp.h[idx];
        hv = hv + DTc * (1.0f - zv) * (nv - hv);
        kp.h[idx] = hv;
        kp.h16[idx] = (_Float16)hv;
      }
    }
    grid_barrier(kp.bar);

    // P5: t1 = relu(h @ pm_w1^T + b)  (p_model #2)
    for (int tt = wid; tt < 256; tt += NWAVES) {
      int mt = tt >> 4, nt = tt & 15;
      v8f c = {};
      c = wmma_part(c, kp.h16, 256, kp.W1, 8, nt, 0, 8, mt * 16);
      int n = nt * 16 + l16;
      float bv = kp.pm_b1[n];
      _Float16* dst = kp.t116 + (size_t)(mt * 16 + lh8) * 256 + n;
      #pragma unroll
      for (int r = 0; r < 8; ++r) {
        float v = c[r] + bv;
        dst[(size_t)r * 256] = (_Float16)(v > 0.0f ? v : 0.0f);
      }
    }
    grid_barrier(kp.bar);

    // P6: pbuf = t1 @ pm_w2^T + b  (f32 only)
    for (int tt = wid; tt < 128; tt += NWAVES) {
      int mt = tt >> 3, nt = tt & 7;
      v8f c = {};
      c = wmma_part(c, kp.t116, 256, kp.W2, 8, nt, 0, 8, mt * 16);
      int n = nt * 16 + l16;
      float bv = kp.pm_b2[n];
      float* df = kp.pbuf + (size_t)(mt * 16 + lh8) * 128 + n;
      #pragma unroll
      for (int r = 0; r < 8; ++r) df[(size_t)r * 128] = c[r] + bv;
    }
    grid_barrier(kp.bar);

    // P7: NLL loss (l1) + observation prep -> gi16 [B][640]
    for (int i = gtid; i < Bb * Dd; i += NTOT) {
      int b = i >> 6, d = i & 63;
      float mean = kp.pbuf[b * 128 + d];
      float lv   = kp.pbuf[b * 128 + 64 + d];
      float x = Xt[i], m = Mt[i];
      float sig = expf(0.5f * lv);
      float err = (x - mean) / sig;
      accL1 += 0.5f * (err * err + lv + LOG2PI) * m;
      const float* w  = kp.wp + d * 40;   // [4][10]
      const float* bb = kp.bp + d * 10;
      _Float16* g = kp.gi16 + (size_t)b * DPc + d * 10;
      #pragma unroll
      for (int j = 0; j < 10; ++j) {
        float a = x * w[j] + mean * w[10 + j] + lv * w[20 + j] + err * w[30 + j] + bb[j];
        a = a > 0.0f ? a : 0.0f;
        g[j] = (_Float16)(a * m);
      }
    }
    grid_barrier(kp.bar);

    // P8: GA = gi @ wih^T + bih  [K=640,N=768];  GB = h @ whh^T + bhh  [K=256,N=768]
    for (int tt = wid; tt < 1536; tt += NWAVES) {
      v8f c = {};
      int mt, nt; float* outp; const float* bias;
      if (tt < 768) {
        mt = tt / 48; nt = tt % 48;
        c = wmma_part(c, kp.gi16, 640, kp.WIH, 20, nt, 0, 20, mt * 16);
        outp = kp.GA; bias = kp.gru_bih;
      } else {
        int t2 = tt - 768; mt = t2 / 48; nt = t2 % 48;
        c = wmma_part(c, kp.h16, 256, kp.WHH, 8, nt, 0, 8, mt * 16);
        outp = kp.GB; bias = kp.gru_bhh;
      }
      int n = nt * 16 + l16;
      float bv = bias[n];
      float* df = outp + (size_t)(mt * 16 + lh8) * 768 + n;
      #pragma unroll
      for (int r = 0; r < 8; ++r) df[(size_t)r * 768] = c[r] + bv;
    }
    grid_barrier(kp.bar);

    // P9: GRU gates (r,z,n order) -> h update
    for (int i = gtid; i < Bb * Hh; i += NTOT) {
      int b = i >> 8, j = i & 255;
      size_t base = (size_t)b * 768;
      float r = sigm(kp.GA[base + j]       + kp.GB[base + j]);
      float z = sigm(kp.GA[base + 256 + j] + kp.GB[base + 256 + j]);
      float n = tanhf(kp.GA[base + 512 + j] + r * kp.GB[base + 512 + j]);
      float hv = (1.0f - z) * n + z * kp.h[i];
      kp.h[i] = hv;
      kp.h16[i] = (_Float16)hv;
    }
    grid_barrier(kp.bar);

    // P10: t1 = relu(h @ pm_w1^T + b)  (p_model #3)
    for (int tt = wid; tt < 256; tt += NWAVES) {
      int mt = tt >> 4, nt = tt & 15;
      v8f c = {};
      c = wmma_part(c, kp.h16, 256, kp.W1, 8, nt, 0, 8, mt * 16);
      int n = nt * 16 + l16;
      float bv = kp.pm_b1[n];
      _Float16* dst = kp.t116 + (size_t)(mt * 16 + lh8) * 256 + n;
      #pragma unroll
      for (int r = 0; r < 8; ++r) {
        float v = c[r] + bv;
        dst[(size_t)r * 256] = (_Float16)(v > 0.0f ? v : 0.0f);
      }
    }
    grid_barrier(kp.bar);

    // P11: pbuf = t1 @ pm_w2^T + b  (consumed by fused KL at next loop top)
    for (int tt = wid; tt < 128; tt += NWAVES) {
      int mt = tt >> 3, nt = tt & 7;
      v8f c = {};
      c = wmma_part(c, kp.t116, 256, kp.W2, 8, nt, 0, 8, mt * 16);
      int n = nt * 16 + l16;
      float bv = kp.pm_b2[n];
      float* df = kp.pbuf + (size_t)(mt * 16 + lh8) * 128 + n;
      #pragma unroll
      for (int r = 0; r < 8; ++r) df[(size_t)r * 128] = c[r] + bv;
    }
    grid_barrier(kp.bar);
  }

  // trailing KL for the last step
  {
    const float* Xp = kp.X  + (size_t)(TSs - 1) * Bb * Dd;
    const float* Mp = kp.Mm + (size_t)(TSs - 1) * Bb * Dd;
    for (int i = gtid; i < Bb * Dd; i += NTOT) {
      int b = i >> 6, d = i & 63;
      float mean2 = kp.pbuf[b * 128 + d];
      float lv2   = kp.pbuf[b * 128 + 64 + d];
      float dx = mean2 - Xp[i];
      float kl = LOG_OBS - 0.5f * lv2 + (expf(lv2) + dx * dx) * INV2VAR - 0.5f;
      accL2 += kl * Mp[i];
    }
  }

  // ---- F1: emit h, classification layer 1, block loss reduction
  for (int i = gtid; i < Bb * Hh; i += NTOT) kp.out[i] = kp.h[i];
  for (int i = gtid; i < Bb * 32; i += NTOT) {           // cls1 = relu(h@cls_w1^T+b1)
    int b = i >> 5, c = i & 31;
    float acc = kp.cls_b1[c];
    for (int k = 0; k < 256; ++k) acc += kp.h[b * 256 + k] * kp.cls_w1[c * 256 + k];
    kp.hid1[i] = acc > 0.0f ? acc : 0.0f;                // reuse hid1 scratch
  }
  sred[tid] = accL1 + accL2;                             // MIXING == 1.0
  __syncthreads();
  for (int s = NTHR / 2; s > 0; s >>= 1) {
    if (tid < s) sred[tid] += sred[tid + s];
    __syncthreads();
  }
  if (tid == 0) kp.partial[blockIdx.x] = sred[0];
  grid_barrier(kp.bar);

  // ---- F2: deterministic final loss sum + classification output
  if (gtid == 0) {
    float s = 0.0f;
    for (int k = 0; k < NBLK; ++k) s += kp.partial[k];
    kp.out[Bb * Hh] = s;
  }
  if (gtid < Bb) {
    float acc = kp.cls_b2[0];
    for (int c = 0; c < 32; ++c) acc += kp.hid1[gtid * 32 + c] * kp.cls_w2[c];
    kp.out[Bb * Hh + 1 + gtid] = acc;
  }
}

// ---------------------------------------------------------------------------
extern "C" void kernel_launch(void* const* d_in, const int* in_sizes, int n_in,
                              void* d_out, int out_size, void* d_ws, size_t ws_size,
                              hipStream_t stream) {
  (void)in_sizes; (void)n_in; (void)out_size; (void)ws_size;
  KParams kp;
  int ii = 0;
  auto nf = [&]() { return (const float*)d_in[ii++]; };
  kp.X = nf();  kp.Mm = nf(); kp.cov = nf();
  kp.cm_w1 = nf(); kp.cm_b1 = nf(); kp.cm_w2 = nf(); kp.cm_b2 = nf();
  kp.pm_w1 = nf(); kp.pm_b1 = nf(); kp.pm_w2 = nf(); kp.pm_b2 = nf();
  kp.gc_wxz = nf(); kp.gc_bxz = nf(); kp.gc_wxn = nf(); kp.gc_bxn = nf();
  kp.gc_whz = nf(); kp.gc_whn = nf();
  kp.wp = nf(); kp.bp = nf();
  kp.gru_wih = nf(); kp.gru_whh = nf(); kp.gru_bih = nf(); kp.gru_bhh = nf();
  kp.cls_w1 = nf(); kp.cls_b1 = nf(); kp.cls_w2 = nf(); kp.cls_b2 = nf();

  char* ws = (char*)d_ws;
  size_t off = 0;
  auto alloc = [&](size_t bytes) {
    void* p = ws + off;
    off = (off + bytes + 255) & ~(size_t)255;
    return p;
  };
  kp.bar     = (int*)  alloc(2 * sizeof(int));
  kp.partial = (float*)alloc(NBLK * sizeof(float));
  kp.W1   = (_Float16*)alloc((size_t)256 * 256 * 2);
  kp.W2   = (_Float16*)alloc((size_t)256 * 128 * 2);
  kp.WZ   = (_Float16*)alloc((size_t)384 * 256 * 2);
  kp.WN   = (_Float16*)alloc((size_t)384 * 256 * 2);
  kp.WIH  = (_Float16*)alloc((size_t)640 * 768 * 2);
  kp.WHH  = (_Float16*)alloc((size_t)256 * 768 * 2);
  kp.h    = (float*)   alloc((size_t)Bb * Hh * 4);
  kp.h16  = (_Float16*)alloc((size_t)Bb * Hh * 2);
  kp.t116 = (_Float16*)alloc((size_t)Bb * 256 * 2);
  kp.p16  = (_Float16*)alloc((size_t)Bb * 128 * 2);
  kp.zh16 = (_Float16*)alloc((size_t)Bb * Hh * 2);
  kp.gi16 = (_Float16*)alloc((size_t)Bb * DPc * 2);
  kp.pbuf = (float*)   alloc((size_t)Bb * 128 * 4);
  kp.zg   = (float*)   alloc((size_t)Bb * Hh * 4);
  kp.GA   = (float*)   alloc((size_t)Bb * 768 * 4);
  kp.GB   = (float*)   alloc((size_t)Bb * 768 * 4);
  kp.hid1 = (float*)   alloc((size_t)Bb * 64 * 4);
  kp.out  = (float*)d_out;

  nnfo_init<<<1, 1, 0, stream>>>(kp.bar);
  nnfo_persistent<<<NBLK, NTHR, 0, stream>>>(kp);
}
